// EdgePredictor_35450660061439
// MI455X (gfx1250) — matrix operand order repaired and checked by reading, compiled, not run on hardware
//
#include <hip/hip_runtime.h>
#include <hip/hip_bf16.h>
#include <math.h>

typedef _Float16 f16;
typedef __attribute__((ext_vector_type(16))) _Float16 v16h;
typedef __attribute__((ext_vector_type(8)))  _Float16 v8h;
typedef __attribute__((ext_vector_type(8)))  float    v8f;

// ---------------------------------------------------------------------------
// helpers
// ---------------------------------------------------------------------------
__device__ __forceinline__ v16h make_frag(const f16* p0, const f16* p1) {
    v8h a = *(const v8h*)p0;
    v8h b = *(const v8h*)p1;
    return __builtin_shufflevector(a, b, 0,1,2,3,4,5,6,7,8,9,10,11,12,13,14,15);
}

__device__ __forceinline__ float gelu_exact(float x) {
    return 0.5f * x * (1.0f + erff(x * 0.70710678118654752f));
}

// ---------------------------------------------------------------------------
// f32 -> f16 conversion
// ---------------------------------------------------------------------------
__global__ void cvt_f32_f16(const float* __restrict__ s, f16* __restrict__ d, int n) {
    int i = blockIdx.x * 256 + threadIdx.x;
    if (i < n) d[i] = (f16)s[i];
}

// ---------------------------------------------------------------------------
// x = vertices @ wp^T + bp   (K = 3, trivial)  -> f32 and f16 copies
// 1024 rows x 512 cols
// ---------------------------------------------------------------------------
__global__ void vproj_kernel(const float* __restrict__ verts,
                             const float* __restrict__ wp,
                             const float* __restrict__ bp,
                             float* __restrict__ xf, f16* __restrict__ xh) {
    int idx  = blockIdx.x * 256 + threadIdx.x;       // < 1024*512
    int row  = idx >> 9;
    int hcol = idx & 511;
    const float* v = verts + row * 3;
    float val = v[0]*wp[hcol*3+0] + v[1]*wp[hcol*3+1] + v[2]*wp[hcol*3+2] + bp[hcol];
    xf[idx] = val;
    xh[idx] = (f16)val;
}

// ---------------------------------------------------------------------------
// Generic WMMA GEMM:  out[M,N] = A[M,K](f16) * W[N,K]^T(f16) (+bias)(+residual)
// Block tile 128x128, 256 threads (8 waves), wave tile 64x32.
// M, N multiples of 128; K multiple of 32.
// ---------------------------------------------------------------------------
#define LDSTR 40   // padded LDS row stride in halves

__global__ __launch_bounds__(256)
void gemm_wmma_f16(const f16* __restrict__ A, int lda,
                   const f16* __restrict__ W, int ldw,
                   const float* __restrict__ bias,
                   const float* __restrict__ residual,
                   float* __restrict__ outF32,
                   f16* __restrict__ outF16,
                   int N, int K) {
    __shared__ f16 sA[128 * LDSTR];
    __shared__ f16 sB[128 * LDSTR];

    const int m0   = blockIdx.x * 128;
    const int n0   = blockIdx.y * 128;
    const int tid  = threadIdx.x;
    const int lane = tid & 31;
    const int w    = tid >> 5;
    const int wm   = w >> 2;   // 0..1  (64-row slab)
    const int wn   = w & 3;    // 0..3  (32-col slab)

    v8f acc[4][2];
    const v8f vzero = {0.f,0.f,0.f,0.f,0.f,0.f,0.f,0.f};
    for (int i = 0; i < 4; ++i)
        for (int j = 0; j < 2; ++j)
            acc[i][j] = vzero;

    const int lr = tid >> 1;          // 0..127 : tile row this thread copies
    const int lc = (tid & 1) * 16;    // 0 or 16 halves

    const int am  = lane & 15;
    const int hiL = lane >> 4;        // 0 or 1
    const int akb = hiL ? 8  : 0;     // A-frag K base (halves)
    const int bkb = hiL ? 8  : 0;     // B-frag K base (halves)

    for (int kt = 0; kt < K; kt += 32) {
        // global -> LDS (32B per thread per matrix)
        const uint4* ga = (const uint4*)(A + (size_t)(m0 + lr) * lda + kt + lc);
        uint4* la = (uint4*)(sA + lr * LDSTR + lc);
        la[0] = ga[0]; la[1] = ga[1];
        const uint4* gb = (const uint4*)(W + (size_t)(n0 + lr) * ldw + kt + lc);
        uint4* lb = (uint4*)(sB + lr * LDSTR + lc);
        lb[0] = gb[0]; lb[1] = gb[1];
        __syncthreads();

        v16h afr[4], bfr[2];
        #pragma unroll
        for (int i = 0; i < 4; ++i) {
            const f16* p = sA + (wm * 64 + i * 16 + am) * LDSTR + akb;
            afr[i] = make_frag(p, p + 16);
        }
        #pragma unroll
        for (int j = 0; j < 2; ++j) {
            const f16* p = sB + (wn * 32 + j * 16 + am) * LDSTR + bkb;
            bfr[j] = make_frag(p, p + 16);
        }
        #pragma unroll
        for (int i = 0; i < 4; ++i)
            #pragma unroll
            for (int j = 0; j < 2; ++j)
                acc[i][j] = __builtin_amdgcn_wmma_f32_16x16x32_f16(
                    false, afr[i], false, bfr[j], (short)0, acc[i][j], false, false);
        __syncthreads();
    }

    // epilogue
    const int cbase = n0 + wn * 32 + am;
    const int rbase = m0 + wm * 64 + (hiL ? 8 : 0);
    for (int i = 0; i < 4; ++i) {
        for (int j = 0; j < 2; ++j) {
            const int col = cbase + j * 16;
            const float bv = bias ? bias[col] : 0.0f;
            for (int r = 0; r < 8; ++r) {
                const int row = rbase + i * 16 + r;
                float v = acc[i][j][r] + bv;
                if (residual) v += residual[(size_t)row * N + col];
                if (outF32) outF32[(size_t)row * N + col] = v;
                if (outF16) outF16[(size_t)row * N + col] = (f16)v;
            }
        }
    }
}

// ---------------------------------------------------------------------------
// Attention: one block per (batch, head). 256 threads; thread t owns q-row t.
// Online softmax; K/V tiles in 64KB of fp16 LDS. ctx written as f16 (M,512).
// ---------------------------------------------------------------------------
__global__ __launch_bounds__(256)
void attn_kernel(const float* __restrict__ qkv, f16* __restrict__ ctx) {
    __shared__ f16 sK[256 * 64];
    __shared__ f16 sV[256 * 64];
    const int b = blockIdx.x >> 3;
    const int h = blockIdx.x & 7;
    const int tid = threadIdx.x;
    const float* base = qkv + (size_t)b * 256 * 1536;

    const float* krow = base + (size_t)tid * 1536 + 512  + h * 64;
    const float* vrow = base + (size_t)tid * 1536 + 1024 + h * 64;
    #pragma unroll 8
    for (int d = 0; d < 64; ++d) {
        sK[tid * 64 + d] = (f16)krow[d];
        sV[tid * 64 + d] = (f16)vrow[d];
    }
    __syncthreads();

    float q[64];
    const float* qrow = base + (size_t)tid * 1536 + h * 64;
    #pragma unroll 8
    for (int d = 0; d < 64; ++d) q[d] = qrow[d];

    float m = -1e30f, s = 0.0f;
    float acc[64];
    #pragma unroll 8
    for (int d = 0; d < 64; ++d) acc[d] = 0.0f;

    for (int k = 0; k < 256; ++k) {
        const f16* kr = sK + k * 64;
        float sc = 0.0f;
        #pragma unroll 8
        for (int d = 0; d < 64; ++d) sc += q[d] * (float)kr[d];
        sc *= 0.125f;                      // 1/sqrt(64)
        float mn   = fmaxf(m, sc);
        float corr = expf(m - mn);
        float p    = expf(sc - mn);
        const f16* vr = sV + k * 64;
        #pragma unroll 8
        for (int d = 0; d < 64; ++d) acc[d] = acc[d] * corr + p * (float)vr[d];
        s = s * corr + p;
        m = mn;
    }
    const float inv = 1.0f / s;
    f16* out = ctx + ((size_t)(b * 256 + tid)) * 512 + h * 64;
    #pragma unroll 8
    for (int d = 0; d < 64; ++d) out[d] = (f16)(acc[d] * inv);
}

// ---------------------------------------------------------------------------
// Fused edge MLP: per block = 16 edges of one batch.
//   h1 = gelu(LN(P1[ii] + P2[jj] + b1))          (16 x 512, fp16 in LDS)
//   h2 = h1 @ w2^T                                (WMMA, 16 x 256)
//   out = sigmoid(gelu(LN(h2 + b2)) . w3 + b3)
// ---------------------------------------------------------------------------
__global__ __launch_bounds__(256)
void edge_kernel(const float* __restrict__ P1, const float* __restrict__ P2,
                 const f16* __restrict__ w2h,
                 const float* __restrict__ b1, const float* __restrict__ g1,
                 const float* __restrict__ be1,
                 const float* __restrict__ b2, const float* __restrict__ g2,
                 const float* __restrict__ be2,
                 const float* __restrict__ w3, const float* __restrict__ b3,
                 float* __restrict__ out) {
    __shared__ f16   sH1[16 * 520];   // gelu(LN(h1)) as f16, padded stride
    __shared__ float sH2[16 * 264];   // h2 (pre-bias) f32, padded stride

    const int b     = blockIdx.x / 2040;
    const int ebase = (blockIdx.x % 2040) * 16;
    const int tid   = threadIdx.x;
    const int lane  = tid & 31;
    const int w     = tid >> 5;
    const int am    = lane & 15;
    const int hiL   = lane >> 4;

    // ---- stage 1: build h1 tile (wave w handles rows 2w, 2w+1) ----
    for (int rr = 0; rr < 2; ++rr) {
        const int r = w * 2 + rr;
        const int e = ebase + r;
        // invert triu_indices(256, k=1): off(i) = i*(511-i)/2
        int i = (int)((511.0f - sqrtf(261121.0f - 8.0f * (float)e)) * 0.5f);
        if (i < 0) i = 0;
        while ((i + 1) * (511 - (i + 1)) / 2 <= e) ++i;
        while (i * (511 - i) / 2 > e) --i;
        const int j = i + 1 + (e - i * (511 - i) / 2);

        const float* p1 = P1 + ((size_t)(b * 256 + i)) * 512;
        const float* p2 = P2 + ((size_t)(b * 256 + j)) * 512;
        float vals[16], sum = 0.0f, sq = 0.0f;
        #pragma unroll
        for (int t = 0; t < 16; ++t) {
            const int c = t * 32 + lane;
            const float v = p1[c] + p2[c] + b1[c];
            vals[t] = v; sum += v; sq += v * v;
        }
        #pragma unroll
        for (int off = 16; off > 0; off >>= 1) {
            sum += __shfl_xor(sum, off, 32);
            sq  += __shfl_xor(sq,  off, 32);
        }
        const float mean = sum * (1.0f / 512.0f);
        const float var  = sq * (1.0f / 512.0f) - mean * mean;
        const float rs   = rsqrtf(var + 1e-5f);
        #pragma unroll
        for (int t = 0; t < 16; ++t) {
            const int c = t * 32 + lane;
            const float x = (vals[t] - mean) * rs * g1[c] + be1[c];
            sH1[r * 520 + c] = (f16)gelu_exact(x);
        }
    }
    __syncthreads();

    // ---- stage 2: h2 = h1 @ w2^T via WMMA (wave w -> n-tiles 2w, 2w+1) ----
    v8f acc[2];
    const v8f vzero = {0.f,0.f,0.f,0.f,0.f,0.f,0.f,0.f};
    acc[0] = vzero; acc[1] = vzero;
    const int n0 = (w * 2) * 16 + am;       // B-frag row for j=0
    const int n1 = (w * 2 + 1) * 16 + am;   // B-frag row for j=1
    for (int kt = 0; kt < 16; ++kt) {
        const int akb = kt * 32 + (hiL ? 8 : 0);
        const f16* pa = sH1 + am * 520 + akb;
        const v16h afr = make_frag(pa, pa + 16);
        const int bkb = kt * 32 + (hiL ? 8 : 0);
        const f16* pb0 = w2h + (size_t)n0 * 512 + bkb;
        const f16* pb1 = w2h + (size_t)n1 * 512 + bkb;
        const v16h bf0 = make_frag(pb0, pb0 + 16);
        const v16h bf1 = make_frag(pb1, pb1 + 16);
        acc[0] = __builtin_amdgcn_wmma_f32_16x16x32_f16(false, afr, false, bf0,
                                                        (short)0, acc[0], false, false);
        acc[1] = __builtin_amdgcn_wmma_f32_16x16x32_f16(false, afr, false, bf1,
                                                        (short)0, acc[1], false, false);
    }

    // ---- stage 3: spill h2 tile to LDS ----
    #pragma unroll
    for (int j = 0; j < 2; ++j) {
        const int n = (w * 2 + j) * 16 + am;
        #pragma unroll
        for (int r8 = 0; r8 < 8; ++r8) {
            const int row = r8 + (hiL ? 8 : 0);
            sH2[row * 264 + n] = acc[j][r8];
        }
    }
    __syncthreads();

    // ---- stage 4: LN + gelu + dot(w3) + sigmoid (wave w -> rows 2w, 2w+1) ----
    for (int rr = 0; rr < 2; ++rr) {
        const int r = w * 2 + rr;
        float vals[8], sum = 0.0f, sq = 0.0f;
        #pragma unroll
        for (int t = 0; t < 8; ++t) {
            const int c = t * 32 + lane;
            const float v = sH2[r * 264 + c] + b2[c];
            vals[t] = v; sum += v; sq += v * v;
        }
        #pragma unroll
        for (int off = 16; off > 0; off >>= 1) {
            sum += __shfl_xor(sum, off, 32);
            sq  += __shfl_xor(sq,  off, 32);
        }
        const float mean = sum * (1.0f / 256.0f);
        const float var  = sq * (1.0f / 256.0f) - mean * mean;
        const float rs   = rsqrtf(var + 1e-5f);
        float dot = 0.0f;
        #pragma unroll
        for (int t = 0; t < 8; ++t) {
            const int c = t * 32 + lane;
            const float x = (vals[t] - mean) * rs * g2[c] + be2[c];
            dot += gelu_exact(x) * w3[c];
        }
        #pragma unroll
        for (int off = 16; off > 0; off >>= 1) dot += __shfl_xor(dot, off, 32);
        if (lane == 0) {
            const float logit = dot + b3[0];
            out[(size_t)b * 32640 + ebase + r] = 1.0f / (1.0f + expf(-logit));
        }
    }
}

// ---------------------------------------------------------------------------
// launch
// ---------------------------------------------------------------------------
extern "C" void kernel_launch(void* const* d_in, const int* in_sizes, int n_in,
                              void* d_out, int out_size, void* d_ws, size_t ws_size,
                              hipStream_t stream) {
    (void)in_sizes; (void)n_in; (void)out_size; (void)ws_size;

    const float* vertices = (const float*)d_in[0];
    const float* wp    = (const float*)d_in[1];
    const float* bp    = (const float*)d_in[2];
    const float* in_w  = (const float*)d_in[3];
    const float* in_b  = (const float*)d_in[4];
    const float* out_w = (const float*)d_in[5];
    const float* out_b = (const float*)d_in[6];
    const float* w1    = (const float*)d_in[7];
    const float* b1    = (const float*)d_in[8];
    const float* g1    = (const float*)d_in[9];
    const float* be1   = (const float*)d_in[10];
    const float* w2    = (const float*)d_in[11];
    const float* b2    = (const float*)d_in[12];
    const float* g2    = (const float*)d_in[13];
    const float* be2   = (const float*)d_in[14];
    const float* w3    = (const float*)d_in[15];
    const float* b3    = (const float*)d_in[16];

    // workspace carve-up
    char* p = (char*)d_ws;
    auto take = [&](size_t bytes) { char* r = p; p += (bytes + 255) & ~size_t(255); return r; };
    float* x_f32   = (float*)take(1024u * 512u * 4u);
    float* qkv     = (float*)take(1024u * 1536u * 4u);
    float* P1      = (float*)take(1024u * 512u * 4u);
    float* P2      = (float*)take(1024u * 512u * 4u);
    f16*   x_f16   = (f16*)take(1024u * 512u * 2u);
    f16*   ctx_h   = (f16*)take(1024u * 512u * 2u);
    f16*   xr_h    = (f16*)take(1024u * 512u * 2u);
    f16*   in_w_h  = (f16*)take(1536u * 512u * 2u);
    f16*   out_w_h = (f16*)take(512u * 512u * 2u);
    f16*   w1_h    = (f16*)take(512u * 1024u * 2u);
    f16*   w2_h    = (f16*)take(256u * 512u * 2u);

    // weight conversions (f32 -> f16)
    cvt_f32_f16<<<(1536*512 + 255) / 256, 256, 0, stream>>>(in_w,  in_w_h,  1536*512);
    cvt_f32_f16<<<(512*512  + 255) / 256, 256, 0, stream>>>(out_w, out_w_h, 512*512);
    cvt_f32_f16<<<(512*1024 + 255) / 256, 256, 0, stream>>>(w1,    w1_h,    512*1024);
    cvt_f32_f16<<<(256*512  + 255) / 256, 256, 0, stream>>>(w2,    w2_h,    256*512);

    // x = vertices @ wp^T + bp
    vproj_kernel<<<(1024 * 512) / 256, 256, 0, stream>>>(vertices, wp, bp, x_f32, x_f16);

    // qkv = x @ in_w^T + in_b        (M=1024, N=1536, K=512)
    gemm_wmma_f16<<<dim3(8, 12), 256, 0, stream>>>(
        x_f16, 512, in_w_h, 512, in_b, nullptr, qkv, nullptr, 1536, 512);

    // attention -> ctx (f16)
    attn_kernel<<<32, 256, 0, stream>>>(qkv, ctx_h);

    // xr = x + ctx @ out_w^T + out_b (f16 out)   (M=1024, N=512, K=512)
    gemm_wmma_f16<<<dim3(8, 4), 256, 0, stream>>>(
        ctx_h, 512, out_w_h, 512, out_b, x_f32, nullptr, xr_h, 512, 512);

    // P1 = xr @ w1[:, :512]^T ;  P2 = xr @ w1[:, 512:]^T
    gemm_wmma_f16<<<dim3(8, 4), 256, 0, stream>>>(
        xr_h, 512, w1_h,       1024, nullptr, nullptr, P1, nullptr, 512, 512);
    gemm_wmma_f16<<<dim3(8, 4), 256, 0, stream>>>(
        xr_h, 512, w1_h + 512, 1024, nullptr, nullptr, P2, nullptr, 512, 512);

    // fused edge MLP -> sigmoid probs
    edge_kernel<<<4 * 2040, 256, 0, stream>>>(
        P1, P2, w2_h, b1, g1, be1, b2, g2, be2, w3, b3, (float*)d_out);
}
